// SSD_10617159156029
// MI455X (gfx1250) — matrix operand order repaired and checked by reading, not compiled, not run on hardware
//
#include <hip/hip_runtime.h>
#include <hip/hip_bf16.h>

// ---- problem geometry -------------------------------------------------------
#define BATCH  32
#define SEQ    2048
#define HID    1024
#define MROWS  (BATCH * SEQ)          // 65536 rows
#define NCONF  8
#define NCLS   128                    // 8 anchors * 16 labels
#define NREG   16                     // 8 anchors * 2
#define NPACK  (NCONF + NCLS + NREG)  // 152 packed output columns
#define NT     10                     // ceil(152/16) -> padded to 160 cols
#define KSTEP  32                     // one bf16 WMMA K-step
#define KSUP   128                    // K-superchunk staged per barrier pair
#define NKS    (KSUP / KSTEP)         // 4 WMMA K-steps per superchunk
#define WAVES  8
#define ROWS_PER_WG (WAVES * 16)      // 128 rows per workgroup
#define FRAGS_PER_SUP (NKS * NT * 32) // 1280 lane-fragments staged per superchunk

typedef __bf16 v16bf __attribute__((ext_vector_type(16)));
typedef float  v8f   __attribute__((ext_vector_type(8)));

__global__ __launch_bounds__(256)
void ssd_head_wmma_kernel(const float* __restrict__ hidden,
                          const float* __restrict__ conf_w, const float* __restrict__ conf_b,
                          const float* __restrict__ cls_w,  const float* __restrict__ cls_b,
                          const float* __restrict__ reg_w,  const float* __restrict__ reg_b,
                          float* __restrict__ out_conf,
                          float* __restrict__ out_cls,
                          float* __restrict__ out_reg)
{
    // Staged B-fragments: [kstep][ntile][lane][16 bf16], hi + lo -> 40 KB + 40 KB
    __shared__ __attribute__((aligned(32))) __bf16 lds_hi[FRAGS_PER_SUP * 16];
    __shared__ __attribute__((aligned(32))) __bf16 lds_lo[FRAGS_PER_SUP * 16];

    const int tid   = threadIdx.x;
    const int lane  = tid & 31;
    const int wave  = tid >> 5;

    const int m0    = blockIdx.x * ROWS_PER_WG + wave * 16;  // wave's 16-row M-tile
    const int mrow  = m0 + (lane & 15);
    const int khalf = lane >> 4;                              // 16-bit A/B layout half

    v8f acc[NT] = {};                                         // f32 C/D accumulators

    const float* arow = hidden + (size_t)mrow * HID;

    for (int kb = 0; kb < HID; kb += KSUP) {
        // ---- stage 152 x 128 f32 weight slice as bf16 hi/lo B-fragments -----
        for (int f = tid; f < FRAGS_PER_SUP; f += 256) {
            const int kstep = f / (NT * 32);                  // 0..3
            const int r     = f - kstep * (NT * 32);
            const int ntile = r >> 5;
            const int flane = r & 31;
            const int ncol  = (ntile << 4) | (flane & 15);    // packed output column
            const int fkh   = flane >> 4;                     // K half: 0 -> K0..15, 1 -> K16..31
            const float* wp = nullptr; int wrow = 0;
            if (ncol < NCONF)                { wp = conf_w; wrow = ncol; }
            else if (ncol < NCONF + NCLS)    { wp = cls_w;  wrow = ncol - NCONF; }
            else if (ncol < NPACK)           { wp = reg_w;  wrow = ncol - (NCONF + NCLS); }
            float vals[16];
            if (wp) {
                const float4* p4 =
                    (const float4*)(wp + (size_t)wrow * HID + kb + kstep * KSTEP + fkh * 16);
                #pragma unroll
                for (int q = 0; q < 4; ++q) {
                    float4 v = p4[q];
                    vals[4*q+0] = v.x; vals[4*q+1] = v.y;
                    vals[4*q+2] = v.z; vals[4*q+3] = v.w;
                }
            } else {
                #pragma unroll
                for (int j = 0; j < 16; ++j) vals[j] = 0.0f;  // pad columns 152..159
            }
            #pragma unroll
            for (int j = 0; j < 16; ++j) {
                float  x = vals[j];
                __bf16 h = (__bf16)x;                          // hi part
                lds_hi[f * 16 + j] = h;
                lds_lo[f * 16 + j] = (__bf16)(x - (float)h);   // lo residual
            }
        }
        __syncthreads();

        // prefetch next superchunk of A toward WGP caches (global_prefetch_b8)
        if (kb + KSUP < HID)
            __builtin_prefetch(arow + kb + KSUP + khalf * 8, 0, 1);

        // ---- 4 K-steps x 10 N-tiles x (hi*hi + hi*lo + lo*hi) ---------------
        // Barrier-free region: scheduler is free to hoist all 16 A loads early.
        #pragma unroll
        for (int ks = 0; ks < NKS; ++ks) {
            // A fragment (16x32 bf16 layout), f32 -> hi + lo split
            // lane<16 : j0..15 -> K0..15
            // lane>=16: j0..15 -> K16..31
            const int ka = kb + ks * KSTEP;
            const float4* pa0 = (const float4*)(arow + ka + khalf * 16);
            const float4* pa1 = (const float4*)(arow + ka + khalf * 16 + 8);
            float4 c0 = pa0[0], c1 = pa0[1], c2 = pa1[0], c3 = pa1[1];
            const float av[16] = { c0.x,c0.y,c0.z,c0.w, c1.x,c1.y,c1.z,c1.w,
                                   c2.x,c2.y,c2.z,c2.w, c3.x,c3.y,c3.z,c3.w };
            v16bf a_hi, a_lo;
            #pragma unroll
            for (int j = 0; j < 16; ++j) {
                float  x = av[j];
                __bf16 h = (__bf16)x;
                a_hi[j] = h;
                a_lo[j] = (__bf16)(x - (float)h);
            }

            const int fbase = ks * (NT * 32);
            #pragma unroll
            for (int nt = 0; nt < NT; ++nt) {
                const v16bf b_hi = *(const v16bf*)(lds_hi + (fbase + nt * 32 + lane) * 16);
                const v16bf b_lo = *(const v16bf*)(lds_lo + (fbase + nt * 32 + lane) * 16);
                acc[nt] = __builtin_amdgcn_wmma_f32_16x16x32_bf16(
                              false, a_hi, false, b_hi, (short)0, acc[nt], false, false);
                acc[nt] = __builtin_amdgcn_wmma_f32_16x16x32_bf16(
                              false, a_hi, false, b_lo, (short)0, acc[nt], false, false);
                acc[nt] = __builtin_amdgcn_wmma_f32_16x16x32_bf16(
                              false, a_lo, false, b_hi, (short)0, acc[nt], false, false);
            }
        }
        __syncthreads();   // protect LDS before next superchunk's staging
    }

    // ---- store: C/D layout lane l -> N = l%16, rows M = r + 8*(l>=16) -------
    const int nlo = lane & 15;
    const int mo  = (lane >> 4) * 8;
    #pragma unroll
    for (int nt = 0; nt < NT; ++nt) {
        const int c = nt * 16 + nlo;
        float* op; int ncols; int col; const float* bp;
        if (c < NCONF)              { op = out_conf; ncols = NCONF; col = c;                  bp = conf_b; }
        else if (c < NCONF + NCLS)  { op = out_cls;  ncols = NCLS;  col = c - NCONF;          bp = cls_b;  }
        else if (c < NPACK)         { op = out_reg;  ncols = NREG;  col = c - (NCONF + NCLS); bp = reg_b;  }
        else continue;              // pad columns 152..159
        const float bias = bp[col];
        #pragma unroll
        for (int r = 0; r < 8; ++r) {
            const int m = m0 + mo + r;
            op[(size_t)m * ncols + col] = acc[nt][r] + bias;
        }
    }
}

extern "C" void kernel_launch(void* const* d_in, const int* in_sizes, int n_in,
                              void* d_out, int out_size, void* d_ws, size_t ws_size,
                              hipStream_t stream) {
    (void)in_sizes; (void)n_in; (void)out_size; (void)d_ws; (void)ws_size;

    const float* hidden = (const float*)d_in[0];
    const float* conf_w = (const float*)d_in[1];
    const float* conf_b = (const float*)d_in[2];
    const float* cls_w  = (const float*)d_in[3];
    const float* cls_b  = (const float*)d_in[4];
    const float* reg_w  = (const float*)d_in[5];
    const float* reg_b  = (const float*)d_in[6];

    float* out      = (float*)d_out;
    float* out_conf = out;                                      // (B,S,8)
    float* out_cls  = out_conf + (size_t)MROWS * NCONF;         // (B,S,8,16)
    float* out_reg  = out_cls  + (size_t)MROWS * NCLS;          // (B,S,8,2)

    dim3 grid(MROWS / ROWS_PER_WG);   // 512 workgroups
    dim3 block(256);                  // 8 wave32s
    hipLaunchKernelGGL(ssd_head_wmma_kernel, grid, block, 0, stream,
                       hidden, conf_w, conf_b, cls_w, cls_b, reg_w, reg_b,
                       out_conf, out_cls, out_reg);
}
